// RecurrentGATCoverageDQN_43018392436912
// MI455X (gfx1250) — matrix operand order repaired and checked by reading, not compiled, not run on hardware
//
#include <hip/hip_runtime.h>
#include <hip/hip_bf16.h>

// ---------------------------------------------------------------------------
// RecurrentGATCoverageDQN on MI455X (gfx1250, wave32, WMMA)
//
// Sizes: N=20000 nodes, E=640000 edges, H=128, L=3 GAT layers, JK=512, NA=9.
// Heavy work: per-layer [20001,128]x[128,128] GEMM (v_wmma_f32_16x16x32_f16,
// f32 accumulate) + 680k-edge segment-softmax aggregation (L2-resident
// gather/scatter). Virtual-node segments (20000 edges -> one dst) are
// pre-reduced per wave / per block to avoid depth-20000 same-address L2
// atomic chains.
// ---------------------------------------------------------------------------

#define NN   20000          // real nodes
#define N1   20001          // nodes + virtual node
#define MP   20016          // N1 padded to multiple of 16 (WMMA M tiles)
#define HH   128            // hidden
#define EE   640000         // input edges
#define ET   680000         // EE + 2*NN virtual edges

typedef _Float16 v8h  __attribute__((ext_vector_type(8)));
typedef _Float16 v16h __attribute__((ext_vector_type(16)));
typedef float    v8f  __attribute__((ext_vector_type(8)));

// ---- helpers --------------------------------------------------------------

__device__ __forceinline__ float wred(float v) {
#pragma unroll
  for (int off = 16; off > 0; off >>= 1) v += __shfl_xor(v, off, 32);
  return v;
}
__device__ __forceinline__ float wredmax(float v) {
#pragma unroll
  for (int off = 16; off > 0; off >>= 1) v = fmaxf(v, __shfl_xor(v, off, 32));
  return v;
}

// monotone float <-> uint mapping for atomicMax on floats
__device__ __forceinline__ unsigned f2ord(float f) {
  unsigned b = __float_as_uint(f);
  return (b & 0x80000000u) ? ~b : (b | 0x80000000u);
}
__device__ __forceinline__ float ord2f(unsigned u) {
  unsigned b = (u & 0x80000000u) ? (u & 0x7FFFFFFFu) : ~u;
  return __uint_as_float(b);
}

__device__ __forceinline__ float sigm(float x) { return 1.0f / (1.0f + __expf(-x)); }

// edge e -> (src, dst):  [0,EE): real edges; [EE,EE+NN): vn->i; rest: i->vn
__device__ __forceinline__ void edge_sd(int e, const int* __restrict__ ei,
                                        int& s, int& d) {
  if (e < EE)            { s = ei[e];          d = ei[EE + e]; }
  else if (e < EE + NN)  { s = NN;             d = e - EE;     }
  else                   { s = e - EE - NN;    d = NN;         }
}

// ---- setup kernels --------------------------------------------------------

__global__ void k_zero(float* p, int n) {
  int i = blockIdx.x * blockDim.x + threadIdx.x;
  if (i < n) p[i] = 0.0f;
}

// h0 = x @ W_node + b_node (K=8 -> VALU). Pooled sum pre-reduced per block
// (64 nodes) in registers+LDS, then one atomic per channel per block.
__global__ void k_encoder(const float* __restrict__ x, const float* __restrict__ Wn,
                          const float* __restrict__ bn, float* __restrict__ h,
                          float* __restrict__ pooled) {
  __shared__ float acc[HH];
  int tid = threadIdx.x;           // 256
  int c = tid & 127, half = tid >> 7;
  int base = blockIdx.x * 64;
  float a = 0.0f;
  for (int r = half; r < 64; r += 2) {
    int n = base + r;
    if (n < NN) {
      const float* xr = x + n * 8;
      float s = bn[c];
#pragma unroll
      for (int k = 0; k < 8; k++) s += xr[k] * Wn[k * HH + c];
      h[(size_t)n * HH + c] = s;
      a += s;
    }
  }
  if (half == 0) acc[c] = a;
  __syncthreads();
  if (half == 1) acc[c] += a;
  __syncthreads();
  if (tid < 128) atomicAdd(&pooled[tid], acc[tid]);
}

// vn = relu(mean(h0) @ W_vn1 + b1) @ W_vn2 + b2 ; writes h[NN] row + vnrows[0]
__global__ void k_vn(const float* __restrict__ pooled,
                     const float* __restrict__ W1, const float* __restrict__ b1,
                     const float* __restrict__ W2, const float* __restrict__ b2,
                     float* __restrict__ h, float* __restrict__ vnrows) {
  __shared__ float pm[HH], t[HH];
  int c = threadIdx.x;
  pm[c] = pooled[c] * (1.0f / (float)NN);
  __syncthreads();
  float s = b1[c];
  for (int k = 0; k < HH; k++) s += pm[k] * W1[k * HH + c];
  t[c] = fmaxf(s, 0.0f);
  __syncthreads();
  float v = b2[c];
  for (int k = 0; k < HH; k++) v += t[k] * W2[k * HH + c];
  h[(size_t)NN * HH + c] = v;
  vnrows[c] = v;
}

// h (f32, N1 rows) -> h16 (f16, MP rows, pad rows zeroed)
__global__ void k_h2half(const float* __restrict__ h, _Float16* __restrict__ h16) {
  int i = blockIdx.x * blockDim.x + threadIdx.x;
  if (i >= MP * HH) return;
  h16[i] = (i < N1 * HH) ? (_Float16)h[i] : (_Float16)0.0f;
}

// W [K=128, N=128] f32 row-major -> Wt16 [N][K] f16 (B operand reads rows)
__global__ void k_wt16(const float* __restrict__ W, _Float16* __restrict__ wt) {
  int i = blockIdx.x * blockDim.x + threadIdx.x;
  if (i >= HH * HH) return;
  int n = i >> 7, k = i & 127;
  wt[i] = (_Float16)W[k * HH + n];
}

// wa[j] = sum_k We[j,k]*att_e[k]  (3x128 dot)
__global__ void k_wa(const float* __restrict__ We, const float* __restrict__ atte,
                     float* __restrict__ wa) {
  int j = threadIdx.x;
  if (j < 3) {
    float s = 0.0f;
    for (int k = 0; k < HH; k++) s += We[j * HH + k] * atte[k];
    wa[j] = s;
  }
}

// ---- WMMA GEMM: hw[MP,128] = h16[MP,128] @ Wt16^T  (one wave = 16x16 tile) --
__global__ void k_gemm(const _Float16* __restrict__ A16,   // [MP,128] row-major
                       const _Float16* __restrict__ BT16,  // [128(N),128(K)] row-major
                       float* __restrict__ C) {            // [MP,128]
  int wave = threadIdx.x >> 5;
  int lane = threadIdx.x & 31;
  int m0 = blockIdx.x * 16;
  int n0 = (blockIdx.y * 4 + wave) * 16;
  int hi = lane >> 4;       // lane half: selects K sub-block per ISA layout
  int lr = lane & 15;

  const _Float16* arow = A16 + (size_t)(m0 + lr) * HH;
  const _Float16* brow = BT16 + (size_t)(n0 + lr) * HH;

  v8f c = {};
#pragma unroll
  for (int kb = 0; kb < 128; kb += 32) {
    // A 16x32 frag: lanes 0-15 rows M, K = kb+[0..7] (v0-3) and kb+[16..23] (v4-7);
    // lanes 16-31 same rows, K offset +8.
    int ab = kb + hi * 8;
    v8h alo = *(const v8h*)(arow + ab);
    v8h ahi = *(const v8h*)(arow + ab + 16);
    v16h a, b;
#pragma unroll
    for (int t = 0; t < 8; t++) { a[t] = alo[t]; a[8 + t] = ahi[t]; }
    // B 32x16 frag: same K order as A (lanes 0-15 K=kb+[0..7],[16..23]; lanes 16-31 +8).
    v8h blo = *(const v8h*)(brow + ab);
    v8h bhi = *(const v8h*)(brow + ab + 16);
#pragma unroll
    for (int t = 0; t < 8; t++) { b[t] = blo[t]; b[8 + t] = bhi[t]; }
    c = __builtin_amdgcn_wmma_f32_16x16x32_f16(false, a, false, b,
                                               (short)0, c, false, false);
  }
  // C layout: c[r] = element (m0 + r + 8*hi, n0 + lr)
#pragma unroll
  for (int r = 0; r < 8; r++)
    C[(size_t)(m0 + r + 8 * hi) * HH + n0 + lr] = c[r];
}

// ls[n] = hw[n]·att_s, ld[n] = hw[n]·att_d  (one wave per node)
__global__ void k_dots(const float* __restrict__ hw, const float* __restrict__ atts,
                       const float* __restrict__ attd, float* __restrict__ ls,
                       float* __restrict__ ld) {
  int wid = (blockIdx.x * blockDim.x + threadIdx.x) >> 5;
  int lane = threadIdx.x & 31;
  if (wid >= N1) return;
  const float* row = hw + (size_t)wid * HH;
  float vs = 0.0f, vd = 0.0f;
#pragma unroll
  for (int j = 0; j < 4; j++) {
    int cc = lane + 32 * j;
    float xv = row[cc];
    vs += xv * atts[cc];
    vd += xv * attd[cc];
  }
  vs = wred(vs); vd = wred(vd);
  if (lane == 0) { ls[wid] = vs; ld[wid] = vd; }
}

// segmax := ord(-1e30), segsum := 0, agg[n] := bias (so atomics add on top)
__global__ void k_seginit(unsigned* __restrict__ segmax, float* __restrict__ segsum,
                          float* __restrict__ agg, const float* __restrict__ bias) {
  int i = blockIdx.x * blockDim.x + threadIdx.x;
  if (i < N1 * HH) agg[i] = bias[i & 127];
  if (i < N1) { segmax[i] = f2ord(-1e30f); segsum[i] = 0.0f; }
}

// logit = leaky_relu(ls[s]+ld[d]+ea·wa, 0.2); segment max via ordered atomicMax.
// Waves fully targeting the virtual node (i->vn range is wave-aligned)
// pre-reduce and issue one atomic per wave.
__global__ void k_logits(const int* __restrict__ ei, const float* __restrict__ ea,
                         const float* __restrict__ ls, const float* __restrict__ ld,
                         const float* __restrict__ wa, float* __restrict__ logits,
                         unsigned* __restrict__ segmax) {
  int e = blockIdx.x * blockDim.x + threadIdx.x;
  if (e >= ET) return;
  int lane = threadIdx.x & 31;
  int s, d; edge_sd(e, ei, s, d);
  float edot = (e < EE)
      ? (ea[e * 3] * wa[0] + ea[e * 3 + 1] * wa[1] + ea[e * 3 + 2] * wa[2])
      : 0.5f * wa[0];
  float lg = ls[s] + ld[d] + edot;
  lg = (lg > 0.0f) ? lg : 0.2f * lg;
  logits[e] = lg;
  if (__all(d == NN)) {
    float mx = wredmax(lg);
    if (lane == 0) atomicMax(&segmax[NN], f2ord(mx));
  } else {
    atomicMax(&segmax[d], f2ord(lg));
  }
}

// expv = exp(logit - segmax[d]); segment sum via atomicAdd (in-place on logits).
// Same wave-level pre-reduction for the i->vn range.
__global__ void k_expsum(const int* __restrict__ ei, float* __restrict__ logits,
                         const unsigned* __restrict__ segmax, float* __restrict__ segsum) {
  int e = blockIdx.x * blockDim.x + threadIdx.x;
  if (e >= ET) return;
  int lane = threadIdx.x & 31;
  int s, d; edge_sd(e, ei, s, d);
  float ev = __expf(logits[e] - ord2f(segmax[d]));
  logits[e] = ev;
  if (__all(d == NN)) {
    float sv = wred(ev);
    if (lane == 0) atomicAdd(&segsum[NN], sv);
  } else {
    atomicAdd(&segsum[d], ev);
  }
}

// agg[d] += hw[s]*alpha for e < EE+NN (random dst -> low contention).
// One wave per edge; float4 row loads; 4 atomics per lane.
__global__ void k_agg(const int* __restrict__ ei, const float* __restrict__ expv,
                      const float* __restrict__ segsum, const float* __restrict__ hw,
                      float* __restrict__ agg) {
  int e = (blockIdx.x * blockDim.x + threadIdx.x) >> 5;
  int lane = threadIdx.x & 31;
  if (e >= EE + NN) return;
  int s, d; edge_sd(e, ei, s, d);
  float alpha = expv[e] / (segsum[d] + 1e-16f);
  const float4* row = (const float4*)(hw + (size_t)s * HH);
  float4 v = row[lane];
  float* dst = agg + (size_t)d * HH + lane * 4;
  atomicAdd(dst + 0, v.x * alpha);
  atomicAdd(dst + 1, v.y * alpha);
  atomicAdd(dst + 2, v.z * alpha);
  atomicAdd(dst + 3, v.w * alpha);
}

// Virtual-node aggregation: agg[NN] += sum_i hw[i]*alpha_i, hierarchical:
// 128 rows per block in registers, LDS combine, 128 atomics per block.
__global__ void k_agg_vn(const float* __restrict__ expv, const float* __restrict__ segsum,
                         const float* __restrict__ hw, float* __restrict__ agg) {
  __shared__ float acc[HH];
  int tid = threadIdx.x;           // 256
  int c = tid & 127, half = tid >> 7;
  float inv = 1.0f / (segsum[NN] + 1e-16f);
  int base = blockIdx.x * 128;
  float a = 0.0f;
  for (int r = half; r < 128; r += 2) {
    int i = base + r;
    if (i < NN) a += hw[(size_t)i * HH + c] * (expv[EE + NN + i] * inv);
  }
  if (half == 0) acc[c] = a;
  __syncthreads();
  if (half == 1) acc[c] += a;
  __syncthreads();
  if (tid < 128) atomicAdd(&agg[(size_t)NN * HH + tid], acc[tid]);
}

// h = relu(layernorm(agg)*gamma+beta + h); write h16; capture virtual-node row
__global__ void k_lnres(const float* __restrict__ agg, const float* __restrict__ gamma,
                        const float* __restrict__ beta, float* __restrict__ h,
                        _Float16* __restrict__ h16, float* __restrict__ vnrows,
                        int layer) {
  int wid = (blockIdx.x * blockDim.x + threadIdx.x) >> 5;
  int lane = threadIdx.x & 31;
  if (wid >= N1) return;
  const float* ar = agg + (size_t)wid * HH;
  float a4[4], s = 0.0f, sq = 0.0f;
#pragma unroll
  for (int j = 0; j < 4; j++) {
    a4[j] = ar[lane + 32 * j];
    s += a4[j]; sq += a4[j] * a4[j];
  }
  s = wred(s); sq = wred(sq);
  float m = s * (1.0f / 128.0f);
  float var = sq * (1.0f / 128.0f) - m * m;
  float rstd = rsqrtf(var + 1e-5f);
#pragma unroll
  for (int j = 0; j < 4; j++) {
    int c = lane + 32 * j;
    float hn = (a4[j] - m) * rstd * gamma[c] + beta[c] + h[(size_t)wid * HH + c];
    hn = fmaxf(hn, 0.0f);
    h[(size_t)wid * HH + c] = hn;
    h16[(size_t)wid * HH + c] = (_Float16)hn;
    if (wid == NN) vnrows[(layer + 1) * HH + c] = hn;
  }
}

// ---- head: GRU + proj + LN + dueling MLPs (tiny; one 256-thread block) -----
__global__ void k_head(const float* __restrict__ vnrows, const float* __restrict__ af,
                       const float* __restrict__ W_ih, const float* __restrict__ b_ih,
                       const float* __restrict__ b_hh,
                       const float* __restrict__ W_proj, const float* __restrict__ b_proj,
                       const float* __restrict__ g_proj, const float* __restrict__ bt_proj,
                       const float* __restrict__ W_rp, const float* __restrict__ b_rp,
                       const float* __restrict__ W_ag, const float* __restrict__ b_ag,
                       const float* __restrict__ W_v1, const float* __restrict__ b_v1,
                       const float* __restrict__ W_v2, const float* __restrict__ b_v2,
                       const float* __restrict__ W_a1, const float* __restrict__ b_a1,
                       const float* __restrict__ W_a2, const float* __restrict__ b_a2,
                       float* __restrict__ out) {
  __shared__ float g[512];
  __shared__ float buf0[640];
  __shared__ float buf1[256];
  __shared__ float red[256];
  __shared__ float sc[2];
  __shared__ float aa[9];
  __shared__ float vv;
  int tid = threadIdx.x;

  for (int j = tid; j < 512; j += 256) g[j] = vnrows[j];       // jk[-1] vn rows
  __syncthreads();
  for (int j = tid; j < 384; j += 256) {                       // gx = g @ W_ih^T + b_ih
    float sv = b_ih[j];
    for (int k = 0; k < 512; k++) sv += g[k] * W_ih[j * 512 + k];
    buf0[j] = sv;
  }
  __syncthreads();
  if (tid < 128) {                                             // GRU gates, h_prev = 0
    float r  = sigm(buf0[tid] + b_hh[tid]);
    float z  = sigm(buf0[128 + tid] + b_hh[128 + tid]);
    float nn = tanhf(buf0[256 + tid] + r * b_hh[256 + tid]);
    buf1[tid] = (1.0f - z) * nn;
  }
  __syncthreads();
  float projc = 0.0f;
  if (tid < 128) {                                             // hg @ W_proj + b_proj
    float sv = b_proj[tid];
    for (int k = 0; k < 128; k++) sv += buf1[k] * W_proj[k * 128 + tid];
    projc = sv;
  }
  red[tid] = (tid < 128) ? projc : 0.0f;
  __syncthreads();
  for (int st = 128; st > 0; st >>= 1) { if (tid < st) red[tid] += red[tid + st]; __syncthreads(); }
  if (tid == 0) sc[0] = red[0] * (1.0f / 128.0f);
  __syncthreads();
  float m = sc[0];
  red[tid] = (tid < 128) ? (projc - m) * (projc - m) : 0.0f;
  __syncthreads();
  for (int st = 128; st > 0; st >>= 1) { if (tid < st) red[tid] += red[tid + st]; __syncthreads(); }
  if (tid == 0) sc[1] = red[0] * (1.0f / 128.0f);
  __syncthreads();
  if (tid < 128) {                                             // rec = relu(LN(...))
    float rec = (projc - m) * rsqrtf(sc[1] + 1e-5f) * g_proj[tid] + bt_proj[tid];
    buf1[tid] = fmaxf(rec, 0.0f);
  }
  __syncthreads();
  for (int j = tid; j < 512; j += 256) {                       // comb[:512] = rec @ W_rp + b_rp
    float sv = b_rp[j];
    for (int k = 0; k < 128; k++) sv += buf1[k] * W_rp[k * 512 + j];
    buf0[j] = sv;
  }
  if (tid < 128) {                                             // comb[512:] = relu(af @ W_ag + b_ag)
    float sv = b_ag[tid];
    for (int k = 0; k < 10; k++) sv += af[k] * W_ag[k * 128 + tid];
    buf0[512 + tid] = fmaxf(sv, 0.0f);
  }
  __syncthreads();
  if (tid < 128) {                                             // value branch hidden
    float sv = b_v1[tid];
    for (int k = 0; k < 640; k++) sv += buf0[k] * W_v1[k * 128 + tid];
    buf1[tid] = fmaxf(sv, 0.0f);
  } else {                                                     // advantage branch hidden
    int cc = tid - 128;
    float sv = b_a1[cc];
    for (int k = 0; k < 640; k++) sv += buf0[k] * W_a1[k * 128 + cc];
    buf1[128 + cc] = fmaxf(sv, 0.0f);
  }
  __syncthreads();
  if (tid == 0) {
    float sv = b_v2[0];
    for (int k = 0; k < 128; k++) sv += buf1[k] * W_v2[k];
    vv = sv;
  }
  if (tid < 9) {
    float sv = b_a2[tid];
    for (int k = 0; k < 128; k++) sv += buf1[128 + k] * W_a2[k * 9 + tid];
    aa[tid] = sv;
  }
  __syncthreads();
  if (tid == 0) {
    float am = 0.0f;
    for (int j = 0; j < 9; j++) am += aa[j];
    am *= (1.0f / 9.0f);
    for (int j = 0; j < 9; j++) out[j] = vv + aa[j] - am;
  }
}

// ---------------------------------------------------------------------------

extern "C" void kernel_launch(void* const* d_in, const int* in_sizes, int n_in,
                              void* d_out, int out_size, void* d_ws, size_t ws_size,
                              hipStream_t stream) {
  // ---- inputs (setup_inputs dict order; params flattened in insertion order)
  const float* x      = (const float*)d_in[0];
  const int*   ei     = (const int*)d_in[1];
  const float* ea     = (const float*)d_in[2];
  const float* af     = (const float*)d_in[3];
  const float* W_node = (const float*)d_in[4];
  const float* b_node = (const float*)d_in[5];
  const float* W_vn1  = (const float*)d_in[6];
  const float* b_vn1  = (const float*)d_in[7];
  const float* W_vn2  = (const float*)d_in[8];
  const float* b_vn2  = (const float*)d_in[9];
  // gat layers occupy indices 10..33 (8 tensors per layer)
  const float* W_ih   = (const float*)d_in[34];
  const float* b_ih   = (const float*)d_in[35];
  const float* b_hh   = (const float*)d_in[37];   // W_hh (36) unused: h_prev = 0
  const float* W_proj = (const float*)d_in[38];
  const float* b_proj = (const float*)d_in[39];
  const float* g_proj = (const float*)d_in[40];
  const float* bt_proj= (const float*)d_in[41];
  const float* W_rp   = (const float*)d_in[42];
  const float* b_rp   = (const float*)d_in[43];
  const float* W_ag   = (const float*)d_in[44];
  const float* b_ag   = (const float*)d_in[45];
  const float* W_v1   = (const float*)d_in[46];
  const float* b_v1   = (const float*)d_in[47];
  const float* W_v2   = (const float*)d_in[48];
  const float* b_v2   = (const float*)d_in[49];
  const float* W_a1   = (const float*)d_in[50];
  const float* b_a1   = (const float*)d_in[51];
  const float* W_a2   = (const float*)d_in[52];
  const float* b_a2   = (const float*)d_in[53];

  // ---- workspace carve-up (256B aligned)
  char* w = (char*)d_ws;
  size_t off = 0;
  auto take = [&](size_t bytes) -> void* {
    void* p = w + off;
    off = (off + bytes + 255) & ~(size_t)255;
    return p;
  };
  float*     h      = (float*)take((size_t)MP * HH * 4);
  _Float16*  h16    = (_Float16*)take((size_t)MP * HH * 2);
  float*     hw     = (float*)take((size_t)MP * HH * 4);
  float*     agg    = (float*)take((size_t)N1 * HH * 4);
  float*     ls     = (float*)take((size_t)N1 * 4);
  float*     ld     = (float*)take((size_t)N1 * 4);
  unsigned*  segmax = (unsigned*)take((size_t)N1 * 4);
  float*     segsum = (float*)take((size_t)N1 * 4);
  float*     logits = (float*)take((size_t)ET * 4);
  float*     pooled = (float*)take(HH * 4);
  float*     vnrows = (float*)take(4 * HH * 4);
  float*     wa     = (float*)take(3 * 4);
  _Float16*  wt16   = (_Float16*)take((size_t)HH * HH * 2);
  (void)ws_size; (void)n_in; (void)in_sizes; (void)out_size;

  // ---- encoder + virtual node
  k_zero<<<1, 128, 0, stream>>>(pooled, HH);
  k_encoder<<<(NN + 63) / 64, 256, 0, stream>>>(x, W_node, b_node, h, pooled);
  k_vn<<<1, 128, 0, stream>>>(pooled, W_vn1, b_vn1, W_vn2, b_vn2, h, vnrows);
  k_h2half<<<(MP * HH + 255) / 256, 256, 0, stream>>>(h, h16);

  // ---- 3 GAT layers
  for (int layer = 0; layer < 3; layer++) {
    int base = 10 + 8 * layer;
    const float* W     = (const float*)d_in[base + 0];
    const float* We    = (const float*)d_in[base + 1];
    const float* att_s = (const float*)d_in[base + 2];
    const float* att_d = (const float*)d_in[base + 3];
    const float* att_e = (const float*)d_in[base + 4];
    const float* bb    = (const float*)d_in[base + 5];
    const float* gamma = (const float*)d_in[base + 6];
    const float* beta  = (const float*)d_in[base + 7];

    k_wt16<<<(HH * HH + 255) / 256, 256, 0, stream>>>(W, wt16);
    k_gemm<<<dim3(MP / 16, 2), 128, 0, stream>>>(h16, wt16, hw);
    k_dots<<<(N1 * 32 + 127) / 128, 128, 0, stream>>>(hw, att_s, att_d, ls, ld);
    k_wa<<<1, 32, 0, stream>>>(We, att_e, wa);
    k_seginit<<<(N1 * HH + 255) / 256, 256, 0, stream>>>(segmax, segsum, agg, bb);
    k_logits<<<(ET + 255) / 256, 256, 0, stream>>>(ei, ea, ls, ld, wa, logits, segmax);
    k_expsum<<<(ET + 255) / 256, 256, 0, stream>>>(ei, logits, segmax, segsum);
    k_agg<<<((EE + NN) * 32 + 255) / 256, 256, 0, stream>>>(ei, logits, segsum, hw, agg);
    k_agg_vn<<<(NN + 127) / 128, 256, 0, stream>>>(logits, segsum, hw, agg);
    k_lnres<<<(N1 * 32 + 127) / 128, 128, 0, stream>>>(agg, gamma, beta, h, h16,
                                                       vnrows, layer);
  }

  // ---- head
  k_head<<<1, 256, 0, stream>>>(vnrows, af, W_ih, b_ih, b_hh, W_proj, b_proj,
                                g_proj, bt_proj, W_rp, b_rp, W_ag, b_ag,
                                W_v1, b_v1, W_v2, b_v2, W_a1, b_a1, W_a2, b_a2,
                                (float*)d_out);
}
